// EvoformerBlock_21741124452588
// MI455X (gfx1250) — matrix operand order repaired and checked by reading, not compiled, not run on hardware
//
#include <hip/hip_runtime.h>

typedef _Float16 h16;
typedef __attribute__((ext_vector_type(16))) _Float16 v16h;
typedef __attribute__((ext_vector_type(8)))  _Float16 v8h;
typedef __attribute__((ext_vector_type(8)))  float    v8f;

constexpr int  kN = 128, kL = 192, kCM = 256, kCZ = 128, kH = 8;
constexpr int  kCOP = 32, kCTRI = 128;
constexpr long kNL = (long)kN * kL;   // 24576 msa rows
constexpr long kLL = (long)kL * kL;   // 36864 pair rows

// ---------------------------------------------------------------------------
// WMMA fragment maps used here (contiguous k, wave32, 16x16x32 f16):
//  A (16x32): element (m,k) -> lane (m&15)+((k&16)?16:0), half k&15
//  B (32x16): element (k,n) -> lane (n&15)+((k&16)?16:0), half k&15
//  C/D f32:   lane holds col n=lane&15; vgpr v -> row m=v+((lane&16)?8:0).
// A v8h chunk (m, k0 in {0,8,16,24}) -> lane (m&15)+((k0&16)?16:0),
// half offset k0&8 : contiguous.  B chunk (n, k0) -> lane
// (n&15)+((k0&16)?16:0), half offset k0&8 : contiguous.
// ---------------------------------------------------------------------------
union F16x16 { v16h v; v8h h[2]; };

// ---------------------------------------------------------------------------
// Weight pre-swizzle: W[K,N] f32 -> fragment-order f16, N zero-padded to 64:
//   out[((kk*nt16 + nt)*32 + lane)*16 + hh] = W[kk*32 + (lane&16)+hh, nt*16+(lane&15)]
// ---------------------------------------------------------------------------
__global__ void k_swz_w(const float* W, h16* out, int K, int N, int padN) {
  long idx = (long)blockIdx.x * 256 + threadIdx.x;
  long tot = (long)(K >> 5) * (padN >> 4) * 512;
  if (idx >= tot) return;
  int hh = (int)(idx & 15);
  int lane = (int)((idx >> 4) & 31);
  long tile = idx >> 9;
  int nt = (int)(tile % (padN >> 4));
  int kk = (int)(tile / (padN >> 4));
  int k = (kk << 5) + (lane & 16) + hh;
  int n = (nt << 4) + (lane & 15);
  out[idx] = (n < N) ? (h16)W[(long)k * N + n] : (h16)0.f;
}

// ---------------------------------------------------------------------------
// Batched WMMA GEMM, compile-time specialized:
//   C[b,m,n] = act( sum_k A[b,m,k]*B[b,k,n] + bias[n] )
// Host invariants: A rows contiguous in K, M%64==0, K%32==0.
// BMODE 1 (BW): B pre-swizzled in global (fragment order), read direct, no LDS.
// BMODE 2 (BK): B[n*bc + k] contiguous in K (tri einsum), staged via LDS.
// OUT: 0 = f32 C, 1 = f16 C.
// 64x64 tile / 128-thread block (4 waves, each 32x32 = 2x2 wmma frags).
// ---------------------------------------------------------------------------
struct GemmP {
  const h16* A; long ar, ab;
  const h16* B; long bc, bb;   // BK only
  int nt16;                    // BW only: padded-N / 16
  const float* bias;
  float* Cf; h16* Ch; long cr, cc, cb;
  int M, Nn, K, act;           // act: 0 none, 1 relu
};

template <int BMODE, int OUT>
__global__ __launch_bounds__(128) void gemm_wmma_kernel(GemmP p) {
  __shared__ h16 Asw[4][32][24];   // [m-tile][lane][16 used] 48B lane stride
  __shared__ h16 Bsw[4][32][24];   // [n-tile][lane][16 used] (BMODE 2 only)
  const int lane = threadIdx.x & 31;
  const int wave = threadIdx.x >> 5;
  const int wm = (wave >> 1) * 32, wn = (wave & 1) * 32;
  const int m0 = blockIdx.x * 64, n0 = blockIdx.y * 64;
  const h16* A = p.A + (long)blockIdx.z * p.ab;
  const h16* B = p.B + (long)blockIdx.z * p.bb;
  v8f acc[2][2] = {};

  for (int kc = 0; kc < p.K; kc += 32) {
    const int kk = kc >> 5;
    // ---- A tile -> fragment-order LDS (128-bit stores) ----
#pragma unroll
    for (int t = 0; t < 2; ++t) {
      int e = threadIdx.x + t * 128;          // 256 chunks
      int m = e >> 2, k0 = (e & 3) << 3;
      v8h val = *(const v8h*)&A[(long)(m0 + m) * p.ar + kc + k0];
      *(v8h*)&Asw[m >> 4][(m & 15) + ((k0 & 16) ? 16 : 0)][k0 & 8] = val;
    }
    // ---- B tile (BK): fragment-order LDS ----
    if (BMODE == 2) {
#pragma unroll
      for (int t = 0; t < 2; ++t) {
        int e = threadIdx.x + t * 128;
        int n = e >> 2, k0 = (e & 3) << 3;
        v8h val = *(const v8h*)&B[(long)(n0 + n) * p.bc + kc + k0];
        *(v8h*)&Bsw[n >> 4][(n & 15) + ((k0 & 16) ? 16 : 0)][k0 & 8] = val;
      }
    }
    __syncthreads();

    // Prefetch next K-slab (global_prefetch_b8).
    if (kc + 32 < p.K) {
      __builtin_prefetch(&A[(long)(m0 + (threadIdx.x & 63)) * p.ar + kc + 32], 0, 1);
      if (BMODE == 2)
        __builtin_prefetch(&B[(long)(n0 + (threadIdx.x & 63)) * p.bc + kc + 32], 0, 1);
    }

    F16x16 af[2], bf[2];
#pragma unroll
    for (int ti = 0; ti < 2; ++ti) {
      const h16* src = &Asw[(wave >> 1) * 2 + ti][lane][0];
      af[ti].h[0] = *(const v8h*)src;
      af[ti].h[1] = *(const v8h*)(src + 8);
    }
    if (BMODE == 1) {
#pragma unroll
      for (int tj = 0; tj < 2; ++tj) {
        long base = (((long)kk * p.nt16 + (n0 >> 4) + (wn >> 4) + tj) << 9) + lane * 16;
        bf[tj].h[0] = *(const v8h*)&p.B[base];
        bf[tj].h[1] = *(const v8h*)&p.B[base + 8];
      }
    } else {
#pragma unroll
      for (int tj = 0; tj < 2; ++tj) {
        const h16* src = &Bsw[(wave & 1) * 2 + tj][lane][0];
        bf[tj].h[0] = *(const v8h*)src;
        bf[tj].h[1] = *(const v8h*)(src + 8);
      }
    }

#pragma unroll
    for (int ti = 0; ti < 2; ++ti)
#pragma unroll
      for (int tj = 0; tj < 2; ++tj)
        acc[ti][tj] = __builtin_amdgcn_wmma_f32_16x16x32_f16(
            false, af[ti].v, false, bf[tj].v, (short)0, acc[ti][tj], false, false);
    __syncthreads();
  }

  const long cz = (long)blockIdx.z * p.cb;
#pragma unroll
  for (int ti = 0; ti < 2; ++ti)
#pragma unroll
    for (int tj = 0; tj < 2; ++tj)
#pragma unroll
      for (int vv = 0; vv < 8; ++vv) {
        int m = m0 + wm + ti * 16 + vv + ((lane & 16) ? 8 : 0);
        int n = n0 + wn + tj * 16 + (lane & 15);
        if (n < p.Nn) {
          float y = acc[ti][tj][vv];
          if (p.bias) y += p.bias[n];
          if (p.act == 1) y = fmaxf(y, 0.f);
          long off = cz + (long)m * p.cr + (long)n * p.cc;
          if (OUT == 0) p.Cf[off] = y;
          else          p.Ch[off] = (h16)y;
        }
      }
}

// ---------------------------------------------------------------------------
// LayerNorm over last dim D; optional f16 and f32 outputs. One row per block.
// ---------------------------------------------------------------------------
__global__ __launch_bounds__(256) void ln_kernel(const float* x, const float* g,
                                                 const float* b, h16* yh,
                                                 float* yf, int D) {
  long row = blockIdx.x;
  const float* xr = x + row * D;
  __shared__ float sh[256], sh2[256];
  float s = 0.f, s2 = 0.f;
  for (int i = threadIdx.x; i < D; i += 256) { float v = xr[i]; s += v; s2 += v * v; }
  sh[threadIdx.x] = s; sh2[threadIdx.x] = s2;
  __syncthreads();
  for (int off = 128; off > 0; off >>= 1) {
    if ((int)threadIdx.x < off) {
      sh[threadIdx.x] += sh[threadIdx.x + off];
      sh2[threadIdx.x] += sh2[threadIdx.x + off];
    }
    __syncthreads();
  }
  float mu = sh[0] / D;
  float var = sh2[0] / D - mu * mu;
  float rs = rsqrtf(var + 1e-5f);
  for (int i = threadIdx.x; i < D; i += 256) {
    float v = (xr[i] - mu) * rs * g[i] + b[i];
    if (yh) yh[row * D + i] = (h16)v;
    if (yf) yf[row * D + i] = v;
  }
}

// ---------------------------------------------------------------------------
// Elementwise kernels
// ---------------------------------------------------------------------------
__global__ void k_cvt(const float* x, h16* y, long n) {
  long i = (long)blockIdx.x * 256 + threadIdx.x;
  if (i < n) y[i] = (h16)x[i];
}
__global__ void k_add(float* a, const float* b, long n) {
  long i = (long)blockIdx.x * 256 + threadIdx.x;
  if (i < n) a[i] += b[i];
}
// out = base + sigmoid(g)*y
__global__ void k_gate_res(float* out, const float* base, const float* g,
                           const float* y, long n) {
  long i = (long)blockIdx.x * 256 + threadIdx.x;
  if (i < n) out[i] = base[i] + (1.f / (1.f + __expf(-g[i]))) * y[i];
}
// out(f16) = a * sigmoid(b)
__global__ void k_mul_sig(h16* out, const float* a, const float* b, long n) {
  long i = (long)blockIdx.x * 256 + threadIdx.x;
  if (i < n) out[i] = (h16)(a[i] * (1.f / (1.f + __expf(-b[i]))));
}
// f16 [A,B,C] -> [C,A,B] (swap==0) or [C,B,A] (swap==1); A=B=kL, C=kCTRI
__global__ void perm_llc_cll(const h16* x, h16* y, int swap) {
  long idx = (long)blockIdx.x * 256 + threadIdx.x;
  long tot = kLL * (long)kCTRI;
  if (idx >= tot) return;
  int c = (int)(idx % kCTRI);
  long ij = idx / kCTRI;
  int jb = (int)(ij % kL), ia = (int)(ij / kL);
  long dst = swap ? ((long)c * kL + jb) * kL + ia
                  : ((long)c * kL + ia) * kL + jb;
  y[dst] = x[idx];
}
// f32 [L,L,CZ] -> transpose first two dims
__global__ void transpose_pair(const float* x, float* y) {
  long idx = (long)blockIdx.x * 256 + threadIdx.x;
  long tot = kLL * (long)kCZ;
  if (idx >= tot) return;
  int c = (int)(idx % kCZ);
  long ij = idx / kCZ;
  int j = (int)(ij % kL), i = (int)(ij / kL);
  y[((long)j * kL + i) * kCZ + c] = x[idx];
}

// ---------------------------------------------------------------------------
// OPM outer product: out[(i*L+j), c*32+d] = alpha * sum_s left[s*L+i,c]*right[s*L+j,d]
// grid (L, L), 128 threads (4 waves = 2x2 of 16x16), K = N = 128 in 4 slabs.
// Slabs staged into LDS in WMMA fragment order: scalar stores on the write
// side (channel dim is the M/N direction here), 2x ds_load_b128 per fragment.
// ---------------------------------------------------------------------------
__global__ __launch_bounds__(128) void opm_outer(const h16* left, const h16* right,
                                                 h16* out, float alpha) {
  __shared__ h16 Lf[2][32][16];   // A-fragment order, c-tile major
  __shared__ h16 Rf[2][32][16];   // B-fragment order, d-tile major
  const int i = blockIdx.x, j = blockIdx.y;
  const int lane = threadIdx.x & 31, wave = threadIdx.x >> 5;
  v8f acc = {};
  for (int kc = 0; kc < kN; kc += 32) {
    {
      int t = threadIdx.x;           // 32 rows x 4 v8h chunks == 128 threads
      int r = t >> 2, c0 = (t & 3) << 3;
      v8h lv = *(const v8h*)&left[((long)(kc + r) * kL + i) * kCOP + c0];
      v8h rv = *(const v8h*)&right[((long)(kc + r) * kL + j) * kCOP + c0];
      int ct = c0 >> 4;
      int lhalf = r & 15;
      int lbase = (r & 16) ? 16 : 0;
      int rbase = (r & 16) ? 16 : 0;
      int rhalf = r & 15;
#pragma unroll
      for (int q = 0; q < 8; ++q) {
        Lf[ct][((c0 + q) & 15) + lbase][lhalf] = lv[q];
        Rf[ct][((c0 + q) & 15) + rbase][rhalf] = rv[q];
      }
    }
    __syncthreads();
    F16x16 a, b;
    a.h[0] = *(const v8h*)&Lf[wave >> 1][lane][0];
    a.h[1] = *(const v8h*)&Lf[wave >> 1][lane][8];
    b.h[0] = *(const v8h*)&Rf[wave & 1][lane][0];
    b.h[1] = *(const v8h*)&Rf[wave & 1][lane][8];
    acc = __builtin_amdgcn_wmma_f32_16x16x32_f16(false, a.v, false, b.v, (short)0,
                                                 acc, false, false);
    __syncthreads();
  }
  const int c0 = (wave >> 1) * 16, d0 = (wave & 1) * 16;
#pragma unroll
  for (int vv = 0; vv < 8; ++vv) {
    int c = c0 + vv + ((lane & 16) ? 8 : 0);
    int d = d0 + (lane & 15);
    out[((long)i * kL + j) * (kCOP * kCOP) + c * kCOP + d] = (h16)(acc[vv] * alpha);
  }
}

// ---------------------------------------------------------------------------
// Flash-lite attention, head dim D (32 for MSA rows, 16 for triangular).
// q/k/v/out: f16 [B*L, H*D], row = b*L + pos, col = h*D + d.
// bias: f32 [(qpos*L + kpos)*H + h], added after 1/sqrt(D) scaling.
// grid (L/16, H, B), 1 wave per block. Whole score row (192) lives in LDS.
// Q, K and the softmax probabilities are kept in WMMA fragment order in LDS
// so every fragment assembles with two ds_load_b128 (V stays a gather).
// ---------------------------------------------------------------------------
template <int D>
__global__ __launch_bounds__(32) void attn_kernel(const h16* q, const h16* k,
                                                  const h16* v, const float* bias,
                                                  h16* out) {
  constexpr int C = kH * D;
  constexpr int CH8 = D / 8;      // v8h chunks per row
  const int lane = threadIdx.x & 31;
  const int b = blockIdx.z, h = blockIdx.y, i0 = blockIdx.x * 16;
  const float scale = rsqrtf((float)D);
  __shared__ float S[16][kL];
  __shared__ h16 Qf[32][16];            // A-fragment order
  __shared__ h16 Kf[32][16];            // B-fragment order
  __shared__ h16 Pf[kL / 32][32][16];   // A-fragment order per K-chunk
  __shared__ h16 Vs[32][D + 8];
  const h16* qb = q + (long)b * kL * C + h * D;
  const h16* kb_ = k + (long)b * kL * C + h * D;
  const h16* vb_ = v + (long)b * kL * C + h * D;
  const v8h vzero = {};

  // ---- stage Q in fragment order ----
  if (D == 16 && lane >= 16) { *(v8h*)&Qf[lane][0] = vzero; *(v8h*)&Qf[lane][8] = vzero; }  // k=16..31 absent
  for (int t = lane; t < 16 * CH8; t += 32) {
    int r = t / CH8, d0 = (t % CH8) * 8;
    v8h val = *(const v8h*)&qb[(long)(i0 + r) * C + d0];
    *(v8h*)&Qf[(r & 15) + ((d0 & 16) ? 16 : 0)][d0 & 8] = val;
  }
  __syncthreads();
  F16x16 qa;
  qa.h[0] = *(const v8h*)&Qf[lane][0];
  qa.h[1] = *(const v8h*)&Qf[lane][8];

  // ---- scores: S[i, j] for all j ----
  for (int j0 = 0; j0 < kL; j0 += 16) {
    if (D == 16 && lane >= 16) {                      // k=16..31 absent
      *(v8h*)&Kf[lane][0] = vzero;
      *(v8h*)&Kf[lane][8] = vzero;
    }
    for (int t = lane; t < 16 * CH8; t += 32) {
      int r = t / CH8, d0 = (t % CH8) * 8;
      v8h val = *(const v8h*)&kb_[(long)(j0 + r) * C + d0];
      *(v8h*)&Kf[(r & 15) + ((d0 & 16) ? 16 : 0)][d0 & 8] = val;
    }
    __syncthreads();
    F16x16 kf;
    kf.h[0] = *(const v8h*)&Kf[lane][0];
    kf.h[1] = *(const v8h*)&Kf[lane][8];
    v8f s = {};
    s = __builtin_amdgcn_wmma_f32_16x16x32_f16(false, qa.v, false, kf.v, (short)0,
                                               s, false, false);
#pragma unroll
    for (int vv = 0; vv < 8; ++vv) {
      int m = vv + ((lane & 16) ? 8 : 0);
      int n = lane & 15;
      S[m][j0 + n] = s[vv] * scale +
                     bias[((long)(i0 + m) * kL + j0 + n) * kH + h];
    }
    __syncthreads();
  }

  // ---- softmax rows, writing probs directly in fragment order ----
  if (lane < 16) {
    float mx = -1e30f;
    for (int j = 0; j < kL; ++j) mx = fmaxf(mx, S[lane][j]);
    float sum = 0.f;
    for (int j = 0; j < kL; ++j) { float e = __expf(S[lane][j] - mx); S[lane][j] = e; sum += e; }
    float inv = 1.f / sum;
    for (int j = 0; j < kL; ++j) {
      int jl = j & 31;
      Pf[j >> 5][lane + (jl & 16)][jl & 15] =
          (h16)(S[lane][j] * inv);
    }
  }
  __syncthreads();

  // ---- P @ V ----
  v8f acc[D / 16] = {};
  for (int kc = 0; kc < kL; kc += 32) {
    for (int t = lane; t < 32 * CH8; t += 32) {
      int r = t / CH8, d0 = (t % CH8) * 8;
      *(v8h*)&Vs[r][d0] = *(const v8h*)&vb_[(long)(kc + r) * C + d0];
    }
    __syncthreads();
    F16x16 pa;
    pa.h[0] = *(const v8h*)&Pf[kc >> 5][lane][0];
    pa.h[1] = *(const v8h*)&Pf[kc >> 5][lane][8];
#pragma unroll
    for (int t = 0; t < D / 16; ++t) {
      v16h vf;
#pragma unroll
      for (int hh = 0; hh < 16; ++hh)
        vf[hh] = Vs[(lane & 16) + hh][t * 16 + (lane & 15)];
      acc[t] = __builtin_amdgcn_wmma_f32_16x16x32_f16(false, pa.v, false, vf,
                                                      (short)0, acc[t], false, false);
    }
    __syncthreads();
  }
#pragma unroll
  for (int t = 0; t < D / 16; ++t)
#pragma unroll
    for (int vv = 0; vv < 8; ++vv) {
      int m = vv + ((lane & 16) ? 8 : 0);
      int n = lane & 15;
      out[((long)b * kL + i0 + m) * C + h * D + t * 16 + n] = (h16)acc[t][vv];
    }
}

// ---------------------------------------------------------------------------
// Host side
// ---------------------------------------------------------------------------
static dim3 ew(long n) { return dim3((unsigned)((n + 255) / 256)); }

// Weight GEMM: A[M,K] f16 x pre-swizzled W -> C[M,Nn] (+bias, act)
static void gemm_w(hipStream_t st, const h16* A, long ar, const h16* Bw,
                   const float* bias, int act, float* Cf, h16* Ch,
                   long cr, int M, int Nn, int K) {
  GemmP p{};
  p.A = A; p.ar = ar; p.ab = 0;
  p.B = Bw; p.bc = 0; p.bb = 0;
  p.nt16 = ((Nn + 63) & ~63) >> 4;
  p.bias = bias; p.Cf = Cf; p.Ch = Ch; p.cr = cr; p.cc = 1; p.cb = 0;
  p.M = M; p.Nn = Nn; p.K = K; p.act = act;
  dim3 g(M / 64, (Nn + 63) / 64, 1);
  if (Cf) gemm_wmma_kernel<1, 0><<<g, dim3(128), 0, st>>>(p);
  else    gemm_wmma_kernel<1, 1><<<g, dim3(128), 0, st>>>(p);
}
// Triangular einsum: batched over kCTRI channels,
// out[i,j,c] = sum_k A[c,i,k] * B[c,j,k]  -> trif[(i*L+j)*CTRI + c] (f32)
static void gemm_tri(hipStream_t st, const h16* A, const h16* B, float* Cf) {
  GemmP p{};
  p.A = A; p.ar = kL; p.ab = kLL;
  p.B = B; p.bc = kL; p.bb = kLL; p.nt16 = 0;
  p.bias = nullptr; p.Cf = Cf; p.Ch = nullptr;
  p.cr = (long)kL * kCTRI; p.cc = kCTRI; p.cb = 1;
  p.M = kL; p.Nn = kL; p.K = kL; p.act = 0;
  dim3 g(kL / 64, kL / 64, kCTRI);
  gemm_wmma_kernel<2, 0><<<g, dim3(128), 0, st>>>(p);
}

// Flat parameter order (dict insertion order, recursive):
enum {
  IN_MSA = 0, IN_PAIR = 1,
  MA_LN_M_G = 2, MA_LN_M_B, MA_LN_Z_G, MA_LN_Z_B, MA_WQ, MA_WK, MA_WV, MA_WPB,
  MA_WO, MA_BO, MA_WG, MA_BG,
  MT_LN_G, MT_LN_B, MT_W1, MT_B1, MT_W2, MT_B2,
  OP_LN_G, OP_LN_B, OP_WL, OP_BL, OP_WR, OP_BR, OP_WOUT, OP_BOUT,
  TO_BASE,                                    // 16 tri_out leaves
  TI_BASE = TO_BASE + 16,                     // 16 tri_in leaves
  TS_BASE = TI_BASE + 16,                     // 10 tri_s leaves
  TE_BASE = TS_BASE + 10,                     // 10 tri_e leaves
  PT_LN_G = TE_BASE + 10, PT_LN_B, PT_W1, PT_B1, PT_W2, PT_B2
};

extern "C" void kernel_launch(void* const* d_in, const int* in_sizes, int n_in,
                              void* d_out, int out_size, void* d_ws,
                              size_t ws_size, hipStream_t stream) {
  (void)in_sizes; (void)n_in; (void)out_size; (void)ws_size;
  auto F = [&](int i) { return (const float*)d_in[i]; };

  char* wp = (char*)d_ws;
  auto alloc = [&](size_t bytes) {
    void* p = wp; wp += (bytes + 255) & ~(size_t)255; return p;
  };
  // Convert + swizzle a weight into WMMA fragment order (padded to N%64).
  auto swz = [&](int idx, int K, int N) {
    int padN = (N + 63) & ~63;
    long tot = (long)(K >> 5) * (padN >> 4) * 512;
    h16* p = (h16*)alloc(tot * sizeof(h16));
    k_swz_w<<<ew(tot), 256, 0, stream>>>(F(idx), p, K, N, padN);
    return p;
  };

  // ---- scratch buffers ----
  float* msa_cur  = (float*)alloc(kNL * kCM * 4);
  float* pair_cur = (float*)alloc(kLL * kCZ * 4);
  float* pair_alt = (float*)alloc(kLL * kCZ * 4);
  h16* lnh  = (h16*)alloc(kNL * kCM * 2);
  h16* qh   = (h16*)alloc(kNL * kCM * 2);
  h16* kh   = (h16*)alloc(kNL * kCM * 2);
  h16* vh   = (h16*)alloc(kNL * kCM * 2);
  h16* aoh  = (h16*)alloc(kNL * kCM * 2);
  h16* hidh = (h16*)alloc(kLL * 1024 * 2);   // hidden / OPM outer (largest f16)
  h16* ltT  = (h16*)alloc((long)kCTRI * kL * kL * 2);
  h16* rtT  = (h16*)alloc((long)kCTRI * kL * kL * 2);
  h16* pch  = (h16*)alloc(kLL * kCZ * 2);
  float* t1 = (float*)alloc(kNL * kCM * 4);
  float* t2 = (float*)alloc(kNL * kCM * 4);
  float* t3 = (float*)alloc(kLL * kCZ * 4);
  float* pb = (float*)alloc(kLL * kH * 4);
  float* trif = (float*)alloc(kLL * kCTRI * 4);

  hipMemcpyAsync(msa_cur, d_in[IN_MSA], kNL * kCM * 4, hipMemcpyDeviceToDevice, stream);
  hipMemcpyAsync(pair_cur, d_in[IN_PAIR], kLL * kCZ * 4, hipMemcpyDeviceToDevice, stream);

  // ================= MSA row attention with pair bias =================
  {
    h16 *Wq = swz(MA_WQ, kCM, kCM), *Wk = swz(MA_WK, kCM, kCM),
        *Wv = swz(MA_WV, kCM, kCM), *Wpb = swz(MA_WPB, kCZ, kH),
        *Wo = swz(MA_WO, kCM, kCM), *Wg = swz(MA_WG, kCM, kCM);
    ln_kernel<<<(unsigned)kNL, 256, 0, stream>>>(msa_cur, F(MA_LN_M_G), F(MA_LN_M_B), lnh, nullptr, kCM);
    ln_kernel<<<(unsigned)kLL, 256, 0, stream>>>(pair_cur, F(MA_LN_Z_G), F(MA_LN_Z_B), pch, nullptr, kCZ);
    gemm_w(stream, lnh, kCM, Wq, nullptr, 0, nullptr, qh, kCM, (int)kNL, kCM, kCM);
    gemm_w(stream, lnh, kCM, Wk, nullptr, 0, nullptr, kh, kCM, (int)kNL, kCM, kCM);
    gemm_w(stream, lnh, kCM, Wv, nullptr, 0, nullptr, vh, kCM, (int)kNL, kCM, kCM);
    gemm_w(stream, lnh, kCM, Wg, F(MA_BG), 0, t1, nullptr, kCM, (int)kNL, kCM, kCM);
    gemm_w(stream, pch, kCZ, Wpb, nullptr, 0, pb, nullptr, kH, (int)kLL, kH, kCZ);
    attn_kernel<32><<<dim3(kL / 16, kH, kN), 32, 0, stream>>>(qh, kh, vh, pb, aoh);
    gemm_w(stream, aoh, kCM, Wo, F(MA_BO), 0, t2, nullptr, kCM, (int)kNL, kCM, kCM);
    k_gate_res<<<ew(kNL * kCM), 256, 0, stream>>>(msa_cur, msa_cur, t1, t2, kNL * kCM);
  }

  // ================= MSA transition =================
  {
    h16 *W1 = swz(MT_W1, kCM, 4 * kCM), *W2 = swz(MT_W2, 4 * kCM, kCM);
    ln_kernel<<<(unsigned)kNL, 256, 0, stream>>>(msa_cur, F(MT_LN_G), F(MT_LN_B), lnh, nullptr, kCM);
    gemm_w(stream, lnh, kCM, W1, F(MT_B1), 1, nullptr, hidh, 4 * kCM, (int)kNL, 4 * kCM, kCM);
    gemm_w(stream, hidh, 4 * kCM, W2, F(MT_B2), 0, t1, nullptr, kCM, (int)kNL, kCM, 4 * kCM);
    k_add<<<ew(kNL * kCM), 256, 0, stream>>>(msa_cur, t1, kNL * kCM);
  }

  // ================= Outer product mean =================
  {
    h16 *Wl = swz(OP_WL, kCM, kCOP), *Wr = swz(OP_WR, kCM, kCOP),
        *Wout = swz(OP_WOUT, kCOP * kCOP, kCZ);
    ln_kernel<<<(unsigned)kNL, 256, 0, stream>>>(msa_cur, F(OP_LN_G), F(OP_LN_B), lnh, nullptr, kCM);
    gemm_w(stream, lnh, kCM, Wl, F(OP_BL), 0, nullptr, qh, kCOP, (int)kNL, kCOP, kCM);
    gemm_w(stream, lnh, kCM, Wr, F(OP_BR), 0, nullptr, kh, kCOP, (int)kNL, kCOP, kCM);
    opm_outer<<<dim3(kL, kL), 128, 0, stream>>>(qh, kh, hidh, 1.f / kN);
    gemm_w(stream, hidh, kCOP * kCOP, Wout, F(OP_BOUT), 0, t1, nullptr, kCZ, (int)kLL, kCZ, kCOP * kCOP);
    k_add<<<ew(kLL * kCZ), 256, 0, stream>>>(pair_cur, t1, kLL * kCZ);
  }

  // ================= Triangular multiplicative updates =================
  auto tri_mult = [&](int base, bool outgoing) {
    const float *lng = F(base + 0), *lnb = F(base + 1), *blp = F(base + 3),
                *brp = F(base + 5), *blg = F(base + 7), *brg = F(base + 9),
                *bog = F(base + 11), *bop = F(base + 13), *fng = F(base + 14),
                *fnb = F(base + 15);
    h16 *Wlp = swz(base + 2, kCZ, kCTRI), *Wrp = swz(base + 4, kCZ, kCTRI),
        *Wlg = swz(base + 6, kCZ, kCTRI), *Wrg = swz(base + 8, kCZ, kCTRI),
        *Wog = swz(base + 10, kCZ, kCZ), *Wop = swz(base + 12, kCTRI, kCZ);
    ln_kernel<<<(unsigned)kLL, 256, 0, stream>>>(pair_cur, lng, lnb, lnh, nullptr, kCZ);
    gemm_w(stream, lnh, kCZ, Wlp, blp, 0, t1, nullptr, kCTRI, (int)kLL, kCTRI, kCZ);
    gemm_w(stream, lnh, kCZ, Wlg, blg, 0, t2, nullptr, kCTRI, (int)kLL, kCTRI, kCZ);
    k_mul_sig<<<ew(kLL * kCTRI), 256, 0, stream>>>(qh, t1, t2, kLL * kCTRI);
    gemm_w(stream, lnh, kCZ, Wrp, brp, 0, t1, nullptr, kCTRI, (int)kLL, kCTRI, kCZ);
    gemm_w(stream, lnh, kCZ, Wrg, brg, 0, t2, nullptr, kCTRI, (int)kLL, kCTRI, kCZ);
    k_mul_sig<<<ew(kLL * kCTRI), 256, 0, stream>>>(kh, t1, t2, kLL * kCTRI);
    // gate on raw pair
    k_cvt<<<ew(kLL * kCZ), 256, 0, stream>>>(pair_cur, pch, kLL * kCZ);
    gemm_w(stream, pch, kCZ, Wog, bog, 0, t3, nullptr, kCZ, (int)kLL, kCZ, kCZ);
    // channel-major repack; for incoming also transpose (i,k)->(k,i) so both
    // modes become out[i,j,c] = sum_k A[c,i,k]*B[c,j,k].
    perm_llc_cll<<<ew(kLL * kCTRI), 256, 0, stream>>>(qh, ltT, outgoing ? 0 : 1);
    perm_llc_cll<<<ew(kLL * kCTRI), 256, 0, stream>>>(kh, rtT, outgoing ? 0 : 1);
    gemm_tri(stream, ltT, rtT, trif);
    ln_kernel<<<(unsigned)kLL, 256, 0, stream>>>(trif, fng, fnb, lnh, nullptr, kCTRI);
    gemm_w(stream, lnh, kCTRI, Wop, bop, 0, t1, nullptr, kCZ, (int)kLL, kCZ, kCTRI);
    k_gate_res<<<ew(kLL * kCZ), 256, 0, stream>>>(pair_cur, pair_cur, t3, t1, kLL * kCZ);
  };
  tri_mult(TO_BASE, true);
  tri_mult(TI_BASE, false);

  // ================= Triangular attention =================
  auto tri_attn = [&](float* pbuf, int base) {
    const float *lng = F(base + 0), *lnb = F(base + 1), *bo = F(base + 7),
                *bg = F(base + 9);
    h16 *Wq = swz(base + 2, kCZ, kCZ), *Wk = swz(base + 3, kCZ, kCZ),
        *Wv = swz(base + 4, kCZ, kCZ), *Wb = swz(base + 5, kCZ, kH),
        *Wo = swz(base + 6, kCZ, kCZ), *Wg = swz(base + 8, kCZ, kCZ);
    ln_kernel<<<(unsigned)kLL, 256, 0, stream>>>(pbuf, lng, lnb, lnh, nullptr, kCZ);
    gemm_w(stream, lnh, kCZ, Wq, nullptr, 0, nullptr, qh, kCZ, (int)kLL, kCZ, kCZ);
    gemm_w(stream, lnh, kCZ, Wk, nullptr, 0, nullptr, kh, kCZ, (int)kLL, kCZ, kCZ);
    gemm_w(stream, lnh, kCZ, Wv, nullptr, 0, nullptr, vh, kCZ, (int)kLL, kCZ, kCZ);
    gemm_w(stream, lnh, kCZ, Wb, nullptr, 0, pb, nullptr, kH, (int)kLL, kH, kCZ);
    attn_kernel<16><<<dim3(kL / 16, kH, kL), 32, 0, stream>>>(qh, kh, vh, pb, aoh);
    gemm_w(stream, aoh, kCZ, Wo, bo, 0, t1, nullptr, kCZ, (int)kLL, kCZ, kCZ);
    k_cvt<<<ew(kLL * kCZ), 256, 0, stream>>>(pbuf, pch, kLL * kCZ);
    gemm_w(stream, pch, kCZ, Wg, bg, 0, t2, nullptr, kCZ, (int)kLL, kCZ, kCZ);
    k_gate_res<<<ew(kLL * kCZ), 256, 0, stream>>>(pbuf, pbuf, t2, t1, kLL * kCZ);
  };
  tri_attn(pair_cur, TS_BASE);  // starting node
  transpose_pair<<<ew(kLL * kCZ), 256, 0, stream>>>(pair_cur, pair_alt);
  tri_attn(pair_alt, TE_BASE);  // ending node (on transposed pair)
  transpose_pair<<<ew(kLL * kCZ), 256, 0, stream>>>(pair_alt, pair_cur);

  // ================= Pair transition =================
  {
    h16 *W1 = swz(PT_W1, kCZ, 4 * kCZ), *W2 = swz(PT_W2, 4 * kCZ, kCZ);
    ln_kernel<<<(unsigned)kLL, 256, 0, stream>>>(pair_cur, F(PT_LN_G), F(PT_LN_B), lnh, nullptr, kCZ);
    gemm_w(stream, lnh, kCZ, W1, F(PT_B1), 1, nullptr, hidh, 4 * kCZ, (int)kLL, 4 * kCZ, kCZ);
    gemm_w(stream, hidh, 4 * kCZ, W2, F(PT_B2), 0, t1, nullptr, kCZ, (int)kLL, kCZ, 4 * kCZ);
    k_add<<<ew(kLL * kCZ), 256, 0, stream>>>(pair_cur, t1, kLL * kCZ);
  }

  // ================= outputs: concat(msa, pair) =================
  hipMemcpyAsync(d_out, msa_cur, kNL * kCM * 4, hipMemcpyDeviceToDevice, stream);
  hipMemcpyAsync((float*)d_out + kNL * kCM, pair_cur, kLL * kCZ * 4,
                 hipMemcpyDeviceToDevice, stream);
}
